// MultiHeadAttention_62886911148063
// MI455X (gfx1250) — matrix operand order repaired and checked by reading, not compiled, not run on hardware
//
#include <hip/hip_runtime.h>
#include <hip/hip_bf16.h>

// MI455X / gfx1250, wave32. MHA: B=4,S=2048,E=1024,H=16,D=64.
// fp32 in/out; bf16 v_wmma_f32_16x16x32_bf16 with f32 accumulation.
// All global->LDS staging uses CDNA5 async loads (global_load_async_to_lds_b128,
// ASYNCcnt) with weights pre-transposed so every tile is row-contiguous.

typedef __attribute__((ext_vector_type(16))) __bf16 v16bf;
typedef __attribute__((ext_vector_type(8)))  float  v8f;

#define B_ 4
#define S_ 2048
#define E_ 1024
#define H_ 16
#define D_ 64
#define HD_ (H_ * D_)
#define LDW 72                 // LDS leading dim; 144 B row pitch (16B aligned)
#define NEG_INF (-3.0e38f)

__device__ __forceinline__ v8f wmma_bf16(v16bf a, v16bf b, v8f c) {
  return __builtin_amdgcn_wmma_f32_16x16x32_bf16(
      false, a, false, b, (short)0, c, false, false);
}

// Generic LDS pointer -> 32-bit LDS byte address (flat aperture: low 32 bits).
__device__ __forceinline__ unsigned lds_addr(const void* p) {
  return (unsigned)(size_t)p;
}

// CDNA5 async copy: 16 bytes global -> LDS, tracked by ASYNCcnt.
__device__ __forceinline__ void async_b128(unsigned lds_off, const void* gsrc) {
  asm volatile("global_load_async_to_lds_b128 %0, %1, off"
               :: "v"(lds_off), "v"(gsrc)
               : "memory");
}
__device__ __forceinline__ void wait_async0() {
  asm volatile("s_wait_asynccnt 0x0" ::: "memory");
}

// A fragment: 16x32 bf16 (MxK), row-major source p[m*ld+k].
// Lane->M (l&15); half h: V0..3 K=h*8+2v+j, V4..7 K=16+h*8+2(v-4)+j.
__device__ __forceinline__ v16bf load_a(const __bf16* p, int ld) {
  const int l = threadIdx.x & 31;
  const int m = l & 15, hh = l >> 4;
  v16bf a;
#pragma unroll
  for (int v = 0; v < 8; ++v) {
    const int k = ((v & 4) << 2) + hh * 8 + ((v & 3) << 1);
    a[2 * v]     = p[m * ld + k];
    a[2 * v + 1] = p[m * ld + k + 1];
  }
  return a;
}

// B fragment: 32x16 bf16 (KxN), source N-major: p[n*ld+k] (K contiguous).
__device__ __forceinline__ v16bf load_b_nk(const __bf16* p, int ld) {
  const int l = threadIdx.x & 31;
  const int n = l & 15, hh = l >> 4;
  v16bf b;
#pragma unroll
  for (int v = 0; v < 8; ++v) {
    const int k = ((v & 4) << 2) + hh * 8 + ((v & 3) << 1);
    b[2 * v]     = p[n * ld + k];
    b[2 * v + 1] = p[n * ld + k + 1];
  }
  return b;
}

// ---------------- converts (one-time, bandwidth-trivial) ----------------
__global__ void cvt_kernel(const float* __restrict__ in,
                           __bf16* __restrict__ out, int n) {
  int i = blockIdx.x * blockDim.x + threadIdx.x;
  const int stride = gridDim.x * blockDim.x;
  for (; i < n; i += stride) out[i] = (__bf16)in[i];
}

// [H,E,D] f32 -> [H,D,E] bf16 (N-major for B-fragments)
__global__ void cvt_w_kernel(const float* __restrict__ in,
                             __bf16* __restrict__ out) {
  int i = blockIdx.x * blockDim.x + threadIdx.x;
  const int stride = gridDim.x * blockDim.x;
  for (; i < H_ * E_ * D_; i += stride) {
    const int e = i & (E_ - 1);
    const int d = (i >> 10) & (D_ - 1);
    const int h = i >> 16;
    out[i] = (__bf16)in[((size_t)h * E_ + e) * D_ + d];
  }
}

// [HD,E] f32 -> [E,HD] bf16 (N-major for B-fragments)
__global__ void cvt_wp_kernel(const float* __restrict__ in,
                              __bf16* __restrict__ out) {
  int i = blockIdx.x * blockDim.x + threadIdx.x;
  const int stride = gridDim.x * blockDim.x;
  for (; i < E_ * HD_; i += stride) {
    const int k = i & (HD_ - 1);
    const int n = i >> 10;
    out[i] = (__bf16)in[(size_t)k * E_ + n];
  }
}

// ---------------- QKV projection GEMM ----------------
// grid (S/64, B, 3*H), 256 thr = 8 waves, 2 accum tiles/wave, K-chunks of 64.
// Tiles async-staged: x [m][k], Wt already [d][e] == [n][k].
__global__ __launch_bounds__(256) void qkv_kernel(
    const __bf16* __restrict__ xb,   // [B,S,E]
    const __bf16* __restrict__ wqt,  // [H,D,E]
    const __bf16* __restrict__ wkt,
    const __bf16* __restrict__ wvt,
    __bf16* __restrict__ qo,         // [B,H,S,D]
    __bf16* __restrict__ ko,         // [B,H,S,D]
    __bf16* __restrict__ vto)        // [B,H,D,S]
{
  __shared__ __bf16 xs[64 * LDW];
  __shared__ __bf16 wsld[64 * LDW];

  const int s0 = blockIdx.x * 64;
  const int bb = blockIdx.y;
  const int z  = blockIdx.z;
  const int which = z / H_;
  const int h     = z % H_;
  const __bf16* w =
      (which == 0 ? wqt : which == 1 ? wkt : wvt) + (size_t)h * D_ * E_;

  const int tid = threadIdx.x;
  const int wid = tid >> 5;
  const int mi  = wid >> 1;
  const int ni0 = (wid & 1) * 2;

  const int row = tid >> 2;        // 0..63
  const int seg = (tid & 3) * 16;  // 32B segment within row

  v8f acc[2] = {};

  for (int k0 = 0; k0 < E_; k0 += 64) {
    const __bf16* xsrc = xb + ((size_t)(bb * S_ + s0 + row)) * E_ + k0 + seg;
    const __bf16* wsrc = w + (size_t)row * E_ + k0 + seg;  // row = d
    async_b128(lds_addr(&xs[row * LDW + seg]),       xsrc);
    async_b128(lds_addr(&xs[row * LDW + seg + 8]),   xsrc + 8);
    async_b128(lds_addr(&wsld[row * LDW + seg]),     wsrc);
    async_b128(lds_addr(&wsld[row * LDW + seg + 8]), wsrc + 8);
    wait_async0();
    __syncthreads();

    v16bf a0 = load_a(xs + mi * 16 * LDW, LDW);
    v16bf a1 = load_a(xs + mi * 16 * LDW + 32, LDW);
#pragma unroll
    for (int t = 0; t < 2; ++t) {
      v16bf b0 = load_b_nk(wsld + (ni0 + t) * 16 * LDW, LDW);
      v16bf b1 = load_b_nk(wsld + (ni0 + t) * 16 * LDW + 32, LDW);
      acc[t] = wmma_bf16(a0, b0, acc[t]);
      acc[t] = wmma_bf16(a1, b1, acc[t]);
    }
    __syncthreads();
  }

  const int l = tid & 31;
  const int nloc = l & 15;
  const int mbase = (l >> 4) * 8;
#pragma unroll
  for (int t = 0; t < 2; ++t) {
    const int ng = (ni0 + t) * 16 + nloc;
#pragma unroll
    for (int r = 0; r < 8; ++r) {
      const int s = s0 + mi * 16 + mbase + r;
      const __bf16 val = (__bf16)acc[t][r];
      if (which == 2) {
        vto[((size_t)(bb * H_ + h) * D_ + ng) * S_ + s] = val;
      } else {
        __bf16* dst = (which == 0 ? qo : ko);
        dst[((size_t)(bb * H_ + h) * S_ + s) * D_ + ng] = val;
      }
    }
  }
}

// ---------------- Flash attention (causal, online softmax) ----------------
// grid (S/64, B*H), 128 thr = 4 waves; wave w owns q-rows [q0+16w,+16).
// K chunk [s][d] and V chunk [d][s] async-staged to LDS once per block.
__global__ __launch_bounds__(128) void flash_kernel(
    const __bf16* __restrict__ q,   // [B,H,S,D]
    const __bf16* __restrict__ kk,  // [B,H,S,D]
    const __bf16* __restrict__ vt,  // [B,H,D,S]
    __bf16* __restrict__ ao)        // [B,S,H*D]
{
  __shared__ __bf16 ks[64 * LDW];
  __shared__ __bf16 vs[64 * LDW];
  __shared__ __bf16 plds[4 * 16 * 64];

  const int q0 = blockIdx.x * 64;
  const int bh = blockIdx.y;
  const int bb = bh / H_, h = bh % H_;
  const int tid = threadIdx.x;
  const int wid = tid >> 5;
  const int qw0 = q0 + wid * 16;

  const __bf16* qbase = q  + ((size_t)bh * S_ + qw0) * D_;
  const __bf16* kbase = kk + (size_t)bh * S_ * D_;
  const __bf16* vbase = vt + (size_t)bh * D_ * S_;
  __bf16* pbuf = plds + wid * 16 * 64;

  const int l = tid & 31;
  const int nloc = l & 15;
  const int mbase = (l >> 4) * 8;

  const int srow = tid >> 1;         // 0..63 staging row
  const int sseg = (tid & 1) * 32;   // 64B half-row

  const v16bf aq0 = load_a(qbase, D_);
  const v16bf aq1 = load_a(qbase + 32, D_);

  v8f o[4] = {};
  float mrow[8], lrow[8];
#pragma unroll
  for (int r = 0; r < 8; ++r) { mrow[r] = NEG_INF; lrow[r] = 0.f; }

  const int nchunks = q0 / 64 + 1;
  const float scale = 0.125f;   // D^-0.5

  for (int kt = 0; kt < nchunks; ++kt) {
    const int kc0 = kt * 64;
    const bool diag = (kt == nchunks - 1);

    // async stage K chunk [s_rel][d] and V chunk [d][s_rel]
    {
      const __bf16* ksrc = kbase + (size_t)(kc0 + srow) * D_ + sseg;
      const __bf16* vsrc = vbase + (size_t)srow * S_ + kc0 + sseg;
      unsigned kdst = lds_addr(&ks[srow * LDW + sseg]);
      unsigned vdst = lds_addr(&vs[srow * LDW + sseg]);
#pragma unroll
      for (int j = 0; j < 4; ++j) {
        async_b128(kdst + 16 * j, ksrc + 8 * j);
        async_b128(vdst + 16 * j, vsrc + 8 * j);
      }
    }
    if (kt + 1 < nchunks)  // global_prefetch_b8 toward next chunk
      __builtin_prefetch(kbase + (size_t)(kc0 + 64) * D_, 0, 0);
    wait_async0();
    __syncthreads();

    // S = scale * Q K^T
    v8f sc[4];
#pragma unroll
    for (int n = 0; n < 4; ++n) {
      v8f t = {};
      t = wmma_bf16(aq0, load_b_nk(ks + n * 16 * LDW, LDW), t);
      t = wmma_bf16(aq1, load_b_nk(ks + n * 16 * LDW + 32, LDW), t);
      sc[n] = t;
    }
#pragma unroll
    for (int n = 0; n < 4; ++n)
#pragma unroll
      for (int r = 0; r < 8; ++r) {
        float v = sc[n][r] * scale;
        if (diag) {
          const int col = kc0 + n * 16 + nloc;
          const int row = qw0 + mbase + r;
          if (col > row) v = NEG_INF;
        }
        sc[n][r] = v;
      }

    // online softmax stats (rows 0..7 in lanes 0..15; 8..15 in 16..31)
    float alpha[8];
#pragma unroll
    for (int r = 0; r < 8; ++r) {
      float v = fmaxf(fmaxf(sc[0][r], sc[1][r]), fmaxf(sc[2][r], sc[3][r]));
#pragma unroll
      for (int off = 1; off < 16; off <<= 1)
        v = fmaxf(v, __shfl_xor(v, off, 32));
      const float mnew = fmaxf(mrow[r], v);
      alpha[r] = __expf(mrow[r] - mnew);
      mrow[r] = mnew;
    }

    float rs[8];
#pragma unroll
    for (int r = 0; r < 8; ++r) rs[r] = 0.f;
#pragma unroll
    for (int n = 0; n < 4; ++n)
#pragma unroll
      for (int r = 0; r < 8; ++r) {
        const float p = __expf(sc[n][r] - mrow[r]);
        rs[r] += p;
        pbuf[(mbase + r) * 64 + n * 16 + nloc] = (__bf16)p;
      }
#pragma unroll
    for (int r = 0; r < 8; ++r) {
      float v = rs[r];
#pragma unroll
      for (int off = 1; off < 16; off <<= 1) v += __shfl_xor(v, off, 32);
      lrow[r] = lrow[r] * alpha[r] + v;
    }
#pragma unroll
    for (int n = 0; n < 4; ++n)
#pragma unroll
      for (int r = 0; r < 8; ++r) o[n][r] *= alpha[r];

    // O += P V   (P from wave-private LDS slab; DS in-order per wave)
#pragma unroll
    for (int kc = 0; kc < 2; ++kc) {
      const v16bf ap = load_a(pbuf + kc * 32, 64);
#pragma unroll
      for (int n = 0; n < 4; ++n)
        o[n] = wmma_bf16(ap, load_b_nk(vs + n * 16 * LDW + kc * 32, LDW), o[n]);
    }
    __syncthreads();  // before next chunk overwrites ks/vs
  }

  // epilogue: O /= l, store bf16 [B,S,H*D]
#pragma unroll
  for (int r = 0; r < 8; ++r) {
    const float inv = 1.f / lrow[r];
    const size_t rowoff =
        ((size_t)bb * S_ + qw0 + mbase + r) * (size_t)HD_ + h * D_ + nloc;
#pragma unroll
    for (int n = 0; n < 4; ++n)
      ao[rowoff + n * 16] = (__bf16)(o[n][r] * inv);
  }
}

// ---------------- Output projection GEMM + bias ----------------
// grid (B*S/64, E/64); Wproj pre-transposed to [E,HD] == [n][k].
__global__ __launch_bounds__(256) void proj_kernel(
    const __bf16* __restrict__ ao,   // [B*S, HD]
    const __bf16* __restrict__ wpt,  // [E, HD]
    const float* __restrict__ bias,  // [E]
    float* __restrict__ out)         // [B*S, E]
{
  __shared__ __bf16 as[64 * LDW];
  __shared__ __bf16 wsld[64 * LDW];

  const int r0 = blockIdx.x * 64;
  const int c0 = blockIdx.y * 64;
  const int tid = threadIdx.x;
  const int wid = tid >> 5;
  const int mi  = wid >> 1;
  const int ni0 = (wid & 1) * 2;

  const int row = tid >> 2;
  const int seg = (tid & 3) * 16;

  v8f acc[2] = {};

  for (int k0 = 0; k0 < HD_; k0 += 64) {
    const __bf16* asrc = ao + (size_t)(r0 + row) * HD_ + k0 + seg;
    const __bf16* wsrc = wpt + (size_t)(c0 + row) * HD_ + k0 + seg;
    async_b128(lds_addr(&as[row * LDW + seg]),       asrc);
    async_b128(lds_addr(&as[row * LDW + seg + 8]),   asrc + 8);
    async_b128(lds_addr(&wsld[row * LDW + seg]),     wsrc);
    async_b128(lds_addr(&wsld[row * LDW + seg + 8]), wsrc + 8);
    wait_async0();
    __syncthreads();

    v16bf a0 = load_a(as + mi * 16 * LDW, LDW);
    v16bf a1 = load_a(as + mi * 16 * LDW + 32, LDW);
#pragma unroll
    for (int t = 0; t < 2; ++t) {
      v16bf b0 = load_b_nk(wsld + (ni0 + t) * 16 * LDW, LDW);
      v16bf b1 = load_b_nk(wsld + (ni0 + t) * 16 * LDW + 32, LDW);
      acc[t] = wmma_bf16(a0, b0, acc[t]);
      acc[t] = wmma_bf16(a1, b1, acc[t]);
    }
    __syncthreads();
  }

  const int l = tid & 31;
  const int nloc = l & 15;
  const int mbase = (l >> 4) * 8;
#pragma unroll
  for (int t = 0; t < 2; ++t) {
    const int cg = c0 + (ni0 + t) * 16 + nloc;
    const float bv = bias[cg];
#pragma unroll
    for (int r = 0; r < 8; ++r) {
      const int rg = r0 + mi * 16 + mbase + r;
      out[(size_t)rg * E_ + cg] = acc[t][r] + bv;
    }
  }
}

extern "C" void kernel_launch(void* const* d_in, const int* in_sizes, int n_in,
                              void* d_out, int out_size, void* d_ws,
                              size_t ws_size, hipStream_t stream) {
  const float* x  = (const float*)d_in[0];
  const float* Wq = (const float*)d_in[1];
  const float* Wk = (const float*)d_in[2];
  const float* Wv = (const float*)d_in[3];
  const float* Wp = (const float*)d_in[4];
  const float* bp = (const float*)d_in[5];
  float* out = (float*)d_out;

  __bf16* ws  = (__bf16*)d_ws;
  __bf16* xb  = ws;
  __bf16* wqb = xb  + (size_t)B_ * S_ * E_;     // [H,D,E]
  __bf16* wkb = wqb + (size_t)H_ * E_ * D_;
  __bf16* wvb = wkb + (size_t)H_ * E_ * D_;
  __bf16* wpb = wvb + (size_t)H_ * E_ * D_;     // [E,HD]
  __bf16* qb  = wpb + (size_t)HD_ * E_;
  __bf16* kb  = qb  + (size_t)B_ * H_ * S_ * D_;
  __bf16* vtb = kb  + (size_t)B_ * H_ * S_ * D_;
  __bf16* aob = vtb + (size_t)B_ * H_ * S_ * D_;

  cvt_kernel<<<2048, 256, 0, stream>>>(x, xb, B_ * S_ * E_);
  cvt_w_kernel<<<1024, 256, 0, stream>>>(Wq, wqb);
  cvt_w_kernel<<<1024, 256, 0, stream>>>(Wk, wkb);
  cvt_w_kernel<<<1024, 256, 0, stream>>>(Wv, wvb);
  cvt_wp_kernel<<<1024, 256, 0, stream>>>(Wp, wpb);

  qkv_kernel<<<dim3(S_ / 64, B_, 3 * H_), 256, 0, stream>>>(
      xb, wqb, wkb, wvb, qb, kb, vtb);

  flash_kernel<<<dim3(S_ / 64, B_ * H_), 128, 0, stream>>>(qb, kb, vtb, aob);

  proj_kernel<<<dim3(B_ * S_ / 64, E_ / 64), 256, 0, stream>>>(
      aob, wpb, bp, out);
}
